// NAttention_21225728377169
// MI455X (gfx1250) — matrix operand order repaired and checked by reading, not compiled, not run on hardware
//
#include <hip/hip_runtime.h>

typedef __bf16 bf16;
typedef __attribute__((ext_vector_type(16))) __bf16 v16bf;
typedef __attribute__((ext_vector_type(8)))  __bf16 v8bf;
typedef __attribute__((ext_vector_type(8)))  float  v8f;

#define NH 6
#define HD 32
#define HH 64
#define WW 64
#define CC 192
#define BB 2
#define KK 224   /* padded key-union: 14 rows x 16 cols */
#define TQ 64    /* 8x8 query tile */

// ---- gfx1250 async global->LDS copy (ASYNCcnt path), with fallback ----
#if defined(__has_builtin)
#if __has_builtin(__builtin_amdgcn_global_load_async_to_lds_b128) && \
    __has_builtin(__builtin_amdgcn_s_wait_asynccnt)
#define HAVE_ASYNC_LDS 1
#endif
#endif

#ifdef HAVE_ASYNC_LDS
// builtin signature (from hipcc diagnostic): first param is
// '__attribute__((vector_size(16))) int __device__ *' (AS1), LDS side is AS3.
typedef int v4i __attribute__((vector_size(16)));
typedef __attribute__((address_space(1))) v4i v4i_g;
typedef __attribute__((address_space(3))) v4i v4i_l;
__device__ __forceinline__ void async_copy16(const bf16* g, bf16* l) {
  __builtin_amdgcn_global_load_async_to_lds_b128(
      (v4i_g*)(void*)g, (v4i_l*)(void*)l, 0, 0);
}
__device__ __forceinline__ void async_wait0() {
  __builtin_amdgcn_s_wait_asynccnt(0);
}
#else
__device__ __forceinline__ void async_copy16(const bf16* g, bf16* l) {
  *(v8bf*)l = *(const v8bf*)g;
}
__device__ __forceinline__ void async_wait0() {}
#endif

// ---- WMMA fragment loaders (16x16x32 bf16; A and B both use contiguous k = 16h + s) ----
// A (16x32): lane L<16 holds row L, K 0..15; lane L+16 holds row L, K 16..31
__device__ __forceinline__ v16bf frag_a(const bf16* base, int ld, int lane) {
  int row = lane & 15;
  int kh  = (lane >> 4) & 1;
  const bf16* p = base + row * ld + kh * 16;
  v8bf lo = *(const v8bf*)(p);
  v8bf hi = *(const v8bf*)(p + 8);
  v16bf a;
  #pragma unroll
  for (int i = 0; i < 8; ++i) { a[i] = lo[i]; a[i + 8] = hi[i]; }
  return a;
}
// B (32x16): lane n<16 holds col n, K 0..15; lane n+16 holds col n, K 16..31
__device__ __forceinline__ v16bf frag_b(const bf16* base, int ld, int lane) {
  int col = lane & 15;
  int kh  = (lane >> 4) & 1;
  return *(const v16bf*)(base + col * ld + kh * 16);
}

// ---- f32 -> bf16 conversion ----
__global__ void k_cvt(const float* __restrict__ s, bf16* __restrict__ d, int n) {
  int i = blockIdx.x * blockDim.x + threadIdx.x;
  if (i < n) d[i] = (bf16)s[i];
}

// ---- QKV projection: 1 wave = 16 rows x 64 cols (4 N-tiles share one A-frag) ----
// grid: (512/8, 9) ; wave w of block handles tm = bx*8+w, N group = by (64 wide)
__global__ void k_qkv(const bf16* __restrict__ xb, const bf16* __restrict__ wb,
                      bf16* __restrict__ qo, bf16* __restrict__ ko, bf16* __restrict__ vo) {
  int lane = threadIdx.x & 31;
  int tm   = blockIdx.x * 8 + (threadIdx.x >> 5);
  int ng   = blockIdx.y;                     // 0..8
  v8f acc[4] = {};
  #pragma unroll
  for (int k0 = 0; k0 < CC; k0 += 32) {
    v16bf a = frag_a(xb + (size_t)(tm * 16) * CC + k0, CC, lane);
    #pragma unroll
    for (int s = 0; s < 4; ++s) {
      v16bf b = frag_b(wb + (size_t)((ng * 4 + s) * 16) * CC + k0, CC, lane);
      acc[s] = __builtin_amdgcn_wmma_f32_16x16x32_bf16(false, a, false, b, (short)0,
                                                       acc[s], false, false);
    }
  }
  #pragma unroll
  for (int s = 0; s < 4; ++s) {
    int dbase = (ng * 4 + s) * 16;           // 16-wide tile never crosses a 32-wide head slot
    int which = dbase / CC;
    int head  = (dbase % CC) / HD;
    int e     = (dbase % HD) + (lane & 15);
    bf16* dst = (which == 0) ? qo : (which == 1) ? ko : vo;
    float mul = (which == 0) ? 0.17677669529663687f : 1.0f;  // hd^-0.5 folded into q
    #pragma unroll
    for (int r = 0; r < 8; ++r) {
      int m  = tm * 16 + r + ((lane >> 4) << 3);
      int b_ = m >> 12, hw = m & 4095;
      size_t idx = ((size_t)(b_ * NH + head) * 4096 + hw) * HD + e;
      dst[idx] = (bf16)(acc[s][r] * mul);
    }
  }
}

// ---- Fused neighborhood attention per (b, head, 8x8 tile); grid (64, NH, BB) ----
__global__ void k_natten(const bf16* __restrict__ qg, const bf16* __restrict__ kg,
                         const bf16* __restrict__ vg, const float* __restrict__ rpb,
                         const float* __restrict__ temp, bf16* __restrict__ og) {
  extern __shared__ char smem[];
  bf16*  sQ  = (bf16*)smem;                       //  64*32  bf16 =  4 KB
  bf16*  sK  = (bf16*)(smem + 4096);              // 224*32  bf16 = 14 KB
  bf16*  sVt = (bf16*)(smem + 4096 + 14336);      //  32*224 bf16 = 14 KB
  float* sS  = (float*)(smem + 32768);            //  64*224 f32  = 56 KB
  bf16*  sP  = (bf16*)(smem + 32768 + 57344);     //  64*224 bf16 = 28 KB

  int t    = threadIdx.x;
  int tile = blockIdx.x;
  int head = blockIdx.y;
  int b    = blockIdx.z;
  int i0 = (tile >> 3) << 3;
  int j0 = (tile & 7) << 3;
  int wr0 = (i0 - 3 > 0) ? i0 - 3 : 0;            // union-window origin (clamped)
  int wc0 = (j0 - 3 > 0) ? j0 - 3 : 0;
  size_t plane = (size_t)(b * NH + head) * 4096;

  // stage Q tile (async global->LDS)
  for (int c = t; c < TQ * 4; c += 256) {
    int qm = c >> 2, e0 = (c & 3) * 8;
    int qi = i0 + (qm >> 3), qj = j0 + (qm & 7);
    async_copy16(qg + (plane + qi * WW + qj) * HD + e0, sQ + qm * HD + e0);
  }
  // stage K union (async) and V union transposed (VGPR bounce: scatter)
  for (int c = t; c < KK * 4; c += 256) {
    int u = c >> 2, e0 = (c & 3) * 8;
    int kr = wr0 + (u >> 4); if (kr > 63) kr = 63;   // clamp; masked in softmax
    int kc = wc0 + (u & 15); if (kc > 63) kc = 63;
    size_t src = (plane + kr * WW + kc) * HD + e0;
    async_copy16(kg + src, sK + u * HD + e0);
    v8bf vv = *(const v8bf*)(vg + src);
    #pragma unroll
    for (int i = 0; i < 8; ++i) sVt[(e0 + i) * KK + u] = vv[i];
  }
  async_wait0();
  __syncthreads();

  int wave = t >> 5, lane = t & 31;

  // QK: 4 row-tiles x 14 col-tiles = 56 WMMAs across 8 waves
  for (int tt = wave; tt < 56; tt += 8) {
    int mt = tt / 14, nt = tt % 14;
    v16bf a = frag_a(sQ + mt * 16 * HD, HD, lane);
    v16bf bm = frag_b(sK + nt * 16 * HD, HD, lane);
    v8f c = {};
    c = __builtin_amdgcn_wmma_f32_16x16x32_bf16(false, a, false, bm, (short)0, c, false, false);
    int n = nt * 16 + (lane & 15);
    #pragma unroll
    for (int r = 0; r < 8; ++r) {
      int m = mt * 16 + r + ((lane >> 4) << 3);
      sS[m * KK + n] = c[r];
    }
  }
  __syncthreads();

  // softmax: one thread per query row over its 49 valid keys (+RPB, *temperature)
  if (t < TQ) {
    int qm = t;
    int qi = i0 + (qm >> 3), qj = j0 + (qm & 7);
    int si = qi - 3; if (si < 0) si = 0; if (si > HH - 7) si = HH - 7;
    int sj = qj - 3; if (sj < 0) sj = 0; if (sj > WW - 7) sj = WW - 7;
    int r0 = si - wr0, c0 = sj - wc0;
    float tp = temp[head];
    float vals[49];
    float mx = -3.0e38f;
    #pragma unroll
    for (int dr = 0; dr < 7; ++dr) {
      #pragma unroll
      for (int dc = 0; dc < 7; ++dc) {
        int u = (r0 + dr) * 16 + (c0 + dc);
        float s = sS[qm * KK + u];
        float bias = rpb[(head * 13 + (si + dr - qi + 6)) * 13 + (sj + dc - qj + 6)];
        float x = (s + bias) * tp;
        vals[dr * 7 + dc] = x;
        mx = x > mx ? x : mx;
      }
    }
    float sum = 0.f;
    #pragma unroll
    for (int i = 0; i < 49; ++i) { float e = __expf(vals[i] - mx); vals[i] = e; sum += e; }
    float inv = 1.0f / sum;
    for (int u = 0; u < KK; ++u) sP[qm * KK + u] = (bf16)0.0f;   // mask invalid keys
    #pragma unroll
    for (int dr = 0; dr < 7; ++dr)
      #pragma unroll
      for (int dc = 0; dc < 7; ++dc)
        sP[qm * KK + (r0 + dr) * 16 + (c0 + dc)] = (bf16)(vals[dr * 7 + dc] * inv);
  }
  __syncthreads();

  // AV: 8 output tiles (4 mt x 2 nt), one per wave, 7 K-chunks each
  {
    int mt = wave >> 1, nt = wave & 1;
    v8f acc = {};
    #pragma unroll
    for (int kc = 0; kc < 7; ++kc) {
      v16bf a = frag_a(sP + mt * 16 * KK + kc * 32, KK, lane);
      v16bf bm = frag_b(sVt + nt * 16 * KK + kc * 32, KK, lane);
      acc = __builtin_amdgcn_wmma_f32_16x16x32_bf16(false, a, false, bm, (short)0, acc, false, false);
    }
    int e = nt * 16 + (lane & 15);
    #pragma unroll
    for (int r = 0; r < 8; ++r) {
      int qm = mt * 16 + r + ((lane >> 4) << 3);
      int qi = i0 + (qm >> 3), qj = j0 + (qm & 7);
      size_t idx = ((size_t)((b * HH + qi) * WW + qj)) * CC + head * HD + e;
      og[idx] = (bf16)acc[r];
    }
  }
}

// ---- output projection: 1 wave = 16 rows x 64 cols; grid (512/8, 3); f32 out ----
__global__ void k_proj(const bf16* __restrict__ ob, const bf16* __restrict__ wb,
                       float* __restrict__ y) {
  int lane = threadIdx.x & 31;
  int tm   = blockIdx.x * 8 + (threadIdx.x >> 5);
  int ng   = blockIdx.y;                      // 0..2
  v8f acc[4] = {};
  #pragma unroll
  for (int k0 = 0; k0 < CC; k0 += 32) {
    v16bf a = frag_a(ob + (size_t)(tm * 16) * CC + k0, CC, lane);
    #pragma unroll
    for (int s = 0; s < 4; ++s) {
      v16bf b = frag_b(wb + (size_t)((ng * 4 + s) * 16) * CC + k0, CC, lane);
      acc[s] = __builtin_amdgcn_wmma_f32_16x16x32_bf16(false, a, false, b, (short)0,
                                                       acc[s], false, false);
    }
  }
  #pragma unroll
  for (int s = 0; s < 4; ++s) {
    int n = (ng * 4 + s) * 16 + (lane & 15);
    #pragma unroll
    for (int r = 0; r < 8; ++r) {
      int m = tm * 16 + r + ((lane >> 4) << 3);
      y[(size_t)m * CC + n] = acc[s][r];
    }
  }
}

extern "C" void kernel_launch(void* const* d_in, const int* in_sizes, int n_in,
                              void* d_out, int out_size, void* d_ws, size_t ws_size,
                              hipStream_t stream) {
  const float* x     = (const float*)d_in[0];
  const float* wqkv  = (const float*)d_in[1];
  const float* rpb   = (const float*)d_in[2];
  const float* temp  = (const float*)d_in[3];
  const float* wproj = (const float*)d_in[4];

  char* w = (char*)d_ws;
  bf16* xb     = (bf16*)w; w += (size_t)BB * HH * WW * CC * 2;   // 3.0 MB
  bf16* wqkvb  = (bf16*)w; w += (size_t)3 * CC * CC * 2;         // 216 KB
  bf16* wprojb = (bf16*)w; w += (size_t)CC * CC * 2;             //  72 KB
  bf16* qb     = (bf16*)w; w += (size_t)BB * NH * HH * WW * HD * 2;
  bf16* kb     = (bf16*)w; w += (size_t)BB * NH * HH * WW * HD * 2;
  bf16* vb     = (bf16*)w; w += (size_t)BB * NH * HH * WW * HD * 2;
  bf16* ab     = (bf16*)w; w += (size_t)BB * HH * WW * CC * 2;

  int nx = BB * HH * WW * CC;        // 1,572,864
  int nw = 3 * CC * CC;              // 110,592
  int np = CC * CC;                  // 36,864
  k_cvt<<<(nx + 255) / 256, 256, 0, stream>>>(x, xb, nx);
  k_cvt<<<(nw + 255) / 256, 256, 0, stream>>>(wqkv, wqkvb, nw);
  k_cvt<<<(np + 255) / 256, 256, 0, stream>>>(wproj, wprojb, np);

  // 512 M-tiles, 8 waves/block covering tm; 9 x 64-wide N groups
  k_qkv<<<dim3(512 / 8, 9), 256, 0, stream>>>(xb, wqkvb, qb, kb, vb);

  // (b, head, 8x8 tile); 118,784 B dynamic LDS (< 320 KB/WGP)
  k_natten<<<dim3(64, NH, BB), 256, 118784, stream>>>(qb, kb, vb, rpb, temp, ab);

  // 512 M-tiles; 3 x 64-wide N groups
  k_proj<<<dim3(512 / 8, 3), 256, 0, stream>>>(ab, wprojb, (float*)d_out);
}
